// GAT_Layer_11613591568919
// MI455X (gfx1250) — matrix operand order repaired and checked by reading, not compiled, not run on hardware
//
#include <hip/hip_runtime.h>

// ---------------------------------------------------------------------------
// GAT layer, MI455X (gfx1250). One workgroup per (b,s) graph: 1024 blocks,
// 256 threads (8 wave32). Both GEMMs use v_wmma_f32_16x16x32_f16.
// ---------------------------------------------------------------------------

typedef __attribute__((ext_vector_type(16))) _Float16 v16h;
typedef __attribute__((ext_vector_type(8)))  _Float16 v8h;
typedef __attribute__((ext_vector_type(2)))  _Float16 v2h;
typedef __attribute__((ext_vector_type(8)))  float    v8f;

#define NN        256     // nodes per graph
#define FD        64      // input features
#define OD        64      // output features
#define XS_STRIDE 72      // halfs per row of staged x   (16B-aligned rows)
#define WT_STRIDE 72      // halfs per row of W^T
#define HT_STRIDE 264     // halfs per row of h^T (264*2 = 528B, 16B aligned)
#define AL_STRIDE 264     // halfs per row of alpha stripe

// dynamic-LDS byte offsets (region 0 = x-stage UNION per-wave alpha stripes)
#define OFF_REG0   0                     // max(256*72*2=36864, 8*16*264*2=67584)
#define OFF_WT     67584                 // 64*72*2   = 9216
#define OFF_HT     76800                 // 64*264*2  = 33792
#define OFF_MASK   110592                // 256*8*4   = 8192
#define OFF_AS     118784                // 256*4
#define OFF_AD     119808
#define OFF_MV     120832
#define OFF_ISV    121856
#define OFF_ATTS   122880                // 64*4
#define OFF_ATTD   123136
#define OFF_BIAS   123392
#define SMEM_BYTES 123648                // 120.75 KB -> 2 blocks / WGP possible

__device__ __forceinline__ v16h cat8(v8h lo, v8h hi) {
  return __builtin_shufflevector(lo, hi, 0,1,2,3,4,5,6,7,8,9,10,11,12,13,14,15);
}

__global__ __launch_bounds__(256) void gat_wmma_kernel(
    const float* __restrict__ x,       // [1024, 256, 64]
    const int*   __restrict__ adj,     // [1024, 256, 256]
    const float* __restrict__ W,       // [64, 64]
    const float* __restrict__ att_src, // [64]
    const float* __restrict__ att_dst, // [64]
    const float* __restrict__ bias,    // [64]
    float*       __restrict__ out)     // [1024, 256, 64]
{
  extern __shared__ __align__(16) char smem[];
  _Float16* XS   = (_Float16*)(smem + OFF_REG0);
  _Float16* WT   = (_Float16*)(smem + OFF_WT);
  _Float16* HT   = (_Float16*)(smem + OFF_HT);
  unsigned* MASK = (unsigned*)(smem + OFF_MASK);
  float*    AS   = (float*)(smem + OFF_AS);
  float*    AD   = (float*)(smem + OFF_AD);
  float*    MV   = (float*)(smem + OFF_MV);
  float*    ISV  = (float*)(smem + OFF_ISV);
  float*    ATTS = (float*)(smem + OFF_ATTS);
  float*    ATTD = (float*)(smem + OFF_ATTD);
  float*    BIA  = (float*)(smem + OFF_BIAS);

  const int g    = blockIdx.x;
  const int tid  = threadIdx.x;
  const int lane = tid & 31;
  const int wave = tid >> 5;
  const int l16  = lane & 15;
  const int lhi  = (lane >> 4) & 1;      // 0 for lanes 0-15, 1 for 16-31

  const float* xg   = x   + (size_t)g * (NN * FD);
  const int*   adjg = adj + (size_t)g * (NN * NN);
  float*       outg = out + (size_t)g * (NN * OD);

  // ----- P0: stage constants, W^T (f16) and x (f16) into LDS ---------------
  if (tid < OD) { ATTS[tid] = att_src[tid]; ATTD[tid] = att_dst[tid]; BIA[tid] = bias[tid]; }
  for (int idx = tid; idx < FD * OD; idx += 256) {
    int k = idx >> 6, o = idx & 63;              // W is [k][o] row-major
    WT[o * WT_STRIDE + k] = (_Float16)W[idx];
  }
  {
    const float2* x2 = (const float2*)xg;        // 8192 float2 per graph
    #pragma unroll 4
    for (int it = 0; it < 32; ++it) {
      int idx = it * 256 + tid;                  // coalesced
      float2 v = x2[idx];
      int row = idx >> 5, cp = idx & 31;         // 32 float2 per 64-wide row
      *(v2h*)&XS[row * XS_STRIDE + cp * 2] = (v2h){(_Float16)v.x, (_Float16)v.y};
    }
  }
  __syncthreads();

  // ----- P1: h = x @ W  (WMMA f16, f32 acc), store h^T (f16) to LDS --------
  // 64 tiles (16 i-tiles x 4 o-tiles), 8 per wave, K=64 -> 2 wmma per tile.
  for (int q = 0; q < 8; ++q) {
    int T  = q * 8 + wave;
    int i0 = (T >> 2) * 16, o0 = (T & 3) * 16;
    v8f acc = {};
    #pragma unroll
    for (int kc = 0; kc < 2; ++kc) {
      int k0 = kc * 32;
      // A: x tile, lane holds M=l16, K=[k0+16*lhi .. +15]
      const _Float16* ap = &XS[(i0 + l16) * XS_STRIDE + k0 + lhi * 16];
      v16h a = cat8(*(const v8h*)ap, *(const v8h*)(ap + 8));
      // B: W tile, lane holds N=o0+l16, K=[k0+16*lhi .. +15] (contiguous in WT row)
      const _Float16* bp = &WT[(o0 + l16) * WT_STRIDE + k0 + lhi * 16];
      v16h b = cat8(*(const v8h*)bp, *(const v8h*)(bp + 8));
      acc = __builtin_amdgcn_wmma_f32_16x16x32_f16(false, a, false, b,
                                                   (short)0, acc, false, false);
    }
    // D layout: lane -> N=o0+l16, VGPR r -> M=i0+r+8*lhi. Store to HT[o][i].
    v8h hv;
    #pragma unroll
    for (int r = 0; r < 8; ++r) hv[r] = (_Float16)acc[r];
    *(v8h*)&HT[(o0 + l16) * HT_STRIDE + i0 + lhi * 8] = hv;
  }
  __syncthreads();

  // ----- P1.5: per-node attention scores a_s, a_d --------------------------
  {
    float as = 0.f, ad = 0.f;
    #pragma unroll 8
    for (int o = 0; o < OD; ++o) {
      float hv = (float)HT[o * HT_STRIDE + tid];
      as = fmaf(hv, ATTS[o], as);
      ad = fmaf(hv, ATTD[o], ad);
    }
    AS[tid] = as; AD[tid] = ad;
  }
  __syncthreads();

  // ----- P2: single coalesced adj pass: online softmax stats + mask bits ---
  // thread i reads adj[j*256 + i] (consecutive across threads for each j).
  {
    const float NEG = -3.402823466e38f;
    float ad = AD[tid];
    float m = NEG, s = 0.f;
    unsigned bits = 0;
    #pragma unroll 4
    for (int j = 0; j < NN; ++j) {
      int a   = adjg[j * NN + tid];
      bool mk = (a != 0) || (j == tid);          // neighbors + self-loop
      bits |= (mk ? 1u : 0u) << (j & 31);
      float e = ad + AS[j];
      e = (e > 0.f) ? e : 0.2f * e;              // leaky_relu(0.2)
      e = mk ? e : NEG;                          // masked-fill (matches ref)
      float nm = fmaxf(m, e);
      s = s * __expf(m - nm) + __expf(e - nm);   // online softmax accumulate
      m = nm;
      if ((j & 31) == 31) { MASK[tid * 8 + (j >> 5)] = bits; bits = 0; }
    }
    MV[tid]  = m;
    ISV[tid] = 1.f / s;
  }
  __syncthreads();

  // ----- P3: per wave, two 16-row stripes: build alpha (f16) then WMMA -----
  _Float16* ALw = (_Float16*)(smem + OFF_REG0) + wave * (16 * AL_STRIDE);
  for (int sp = 0; sp < 2; ++sp) {
    int i0 = (wave + sp * 8) * 16;

    // alpha[i0+r][j] = mask ? exp(e - m_i) / s_i : 0, stored f16
    #pragma unroll 4
    for (int it = 0; it < 64; ++it) {
      int r = it >> 2, sub = it & 3;
      int i = i0 + r;
      int j = sub * 64 + lane * 2;               // 2 consecutive j per lane
      unsigned w = MASK[i * 8 + (j >> 5)];       // broadcast-ish LDS read
      float mi = MV[i], isi = ISV[i], adi = AD[i];
      float e0 = adi + AS[j];     e0 = (e0 > 0.f) ? e0 : 0.2f * e0;
      float e1 = adi + AS[j + 1]; e1 = (e1 > 0.f) ? e1 : 0.2f * e1;
      float a0 = ((w >> (j & 31)) & 1u)       ? __expf(e0 - mi) * isi : 0.f;
      float a1 = ((w >> ((j + 1) & 31)) & 1u) ? __expf(e1 - mi) * isi : 0.f;
      *(v2h*)&ALw[r * AL_STRIDE + j] = (v2h){(_Float16)a0, (_Float16)a1};
    }
    // ALw is wave-private; per-wave LDS ops are in-order (DScnt), no barrier.

    // out[i0..i0+15][:] = alpha_stripe @ h   (K = 256 -> 8 wmma per o-tile)
    for (int ot = 0; ot < 4; ++ot) {
      v8f acc = {};
      #pragma unroll
      for (int kc = 0; kc < 8; ++kc) {
        int k0 = kc * 32;
        const _Float16* ap = &ALw[l16 * AL_STRIDE + k0 + lhi * 16];
        v16h a = cat8(*(const v8h*)ap, *(const v8h*)(ap + 8));
        const _Float16* bp = &HT[(ot * 16 + l16) * HT_STRIDE + k0 + lhi * 16];
        v16h b = cat8(*(const v8h*)bp, *(const v8h*)(bp + 8));
        acc = __builtin_amdgcn_wmma_f32_16x16x32_f16(false, a, false, b,
                                                     (short)0, acc, false, false);
      }
      int o = ot * 16 + l16;
      float bo = BIA[o];
      #pragma unroll
      for (int r = 0; r < 8; ++r) {
        int i = i0 + r + lhi * 8;
        outg[i * OD + o] = acc[r] + bo;
      }
    }
  }
}

// ---------------------------------------------------------------------------
extern "C" void kernel_launch(void* const* d_in, const int* in_sizes, int n_in,
                              void* d_out, int out_size, void* d_ws, size_t ws_size,
                              hipStream_t stream) {
  (void)in_sizes; (void)n_in; (void)out_size; (void)d_ws; (void)ws_size;
  const float* x       = (const float*)d_in[0];
  const int*   adj     = (const int*)d_in[1];
  const float* W       = (const float*)d_in[2];
  const float* att_src = (const float*)d_in[3];
  const float* att_dst = (const float*)d_in[4];
  const float* bias    = (const float*)d_in[5];
  float*       out     = (float*)d_out;

  dim3 grid(16 * 64);   // one block per (b,s) graph
  dim3 block(256);      // 8 wave32
  gat_wmma_kernel<<<grid, block, SMEM_BYTES, stream>>>(
      x, adj, W, att_src, att_dst, bias, out);
}
